// VGGSNNServer_6141803233489
// MI455X (gfx1250) — matrix operand order repaired and checked by reading, not compiled, not run on hardware
//
#include <hip/hip_runtime.h>
#include <hip/hip_bf16.h>

// Problem dims (fixed by reference)
#define TSTEPS 20
#define BATCH  512
#define DIN    4096
#define HID    1024
#define NCLS   100
#define LEAK   0.95f
#define EPSBN  1e-4f

typedef __attribute__((ext_vector_type(16))) _Float16 v16h;
typedef __attribute__((ext_vector_type(8)))  _Float16 v8h;
typedef __attribute__((ext_vector_type(4)))  _Float16 v4h;
typedef __attribute__((ext_vector_type(8)))  float    v8f;
typedef __attribute__((ext_vector_type(4)))  float    v4f;

// ---------------------------------------------------------------------------
// fp32 -> f16 converter (W1 once per launch; z[t] once per step).
// 4 elements / thread, b128 in, b64 out.
// ---------------------------------------------------------------------------
__global__ void snn_cvt_f32_to_f16(const float* __restrict__ src,
                                   _Float16* __restrict__ dst) {
    int i = blockIdx.x * blockDim.x + threadIdx.x;
    v4f f = ((const v4f*)src)[i];
    v4h h;
    h[0] = (_Float16)f.x; h[1] = (_Float16)f.y;
    h[2] = (_Float16)f.z; h[3] = (_Float16)f.w;
    ((v4h*)dst)[i] = h;
}

// ---------------------------------------------------------------------------
// Zero persistent state (mem1, mem2) — must run every launch (determinism)
// ---------------------------------------------------------------------------
__global__ void snn_zero_state(float* __restrict__ mem1, float* __restrict__ mem2) {
    int i = blockIdx.x * blockDim.x + threadIdx.x;
    if (i < BATCH * HID)  mem1[i] = 0.0f;
    if (i < BATCH * NCLS) mem2[i] = 0.0f;
}

// ---------------------------------------------------------------------------
// WMMA helpers — layouts per CDNA5 ISA §7.12.2
//   A f16 16x32: lane L<16 -> M=L, K in [0..7]∪[16..23]; lane L+16 -> K [8..15]∪[24..31]
//   B f16 32x16: lane n -> N=n, K same as A (lane<16: [0..7]∪[16..23]; lanes 16-31: [8..15]∪[24..31])
//   C  f32 16x16: VGPR v -> M = v + 8*(lane>=16), N = lane&15
// ---------------------------------------------------------------------------
__device__ __forceinline__ v8f wmma_f16f32(v16h a, v16h b, v8f c) {
    return __builtin_amdgcn_wmma_f32_16x16x32_f16(
        /*neg_a=*/false, a, /*neg_b=*/false, b,
        /*c_mod=*/(short)0, c, /*reuse_a=*/false, /*reuse_b=*/false);
}

// A fragment (f16 source): two 16B chunks at p and p+16 halves.
__device__ __forceinline__ v16h load_a_f16(const _Float16* __restrict__ p) {
    v8h lo = *(const v8h*)p;
    v8h hi = *(const v8h*)(p + 16);
    v16h a;
#pragma unroll
    for (int i = 0; i < 8; ++i) { a[i] = lo[i]; a[8 + i] = hi[i]; }
    return a;
}

// B fragment: two 16B chunks at p and p+16 halves (two b128 loads).
__device__ __forceinline__ v16h load_b_f16(const _Float16* __restrict__ p) {
    v8h lo = *(const v8h*)p;
    v8h hi = *(const v8h*)(p + 16);
    v16h b;
#pragma unroll
    for (int i = 0; i < 8; ++i) { b[i] = lo[i]; b[8 + i] = hi[i]; }
    return b;
}

// ---------------------------------------------------------------------------
// Kernel A: h[B,H] = z16[B,D] @ W1h[H,D]^T  via WMMA f16->f32
// Block: 256 threads (8 waves) = 128M x 128N tile; wave = 64M x 32N
// (4 M-frags x 2 N-frags = 8 accumulators, 8 WMMA per K-step of 32).
// Fragment rows sit at compile-time byte offsets (f*16*DIN*2 = 128KB < 8MB
// IOFFSET range) from TWO base pointers -> no pointer arrays, no spills.
// ---------------------------------------------------------------------------
__global__ void snn_gemm1_wmma(const _Float16* __restrict__ z16,  // [BATCH, DIN]
                               const _Float16* __restrict__ w1h,  // [HID, DIN]
                               float* __restrict__ h_out)         // [BATCH, HID]
{
    const int lane = threadIdx.x & 31;
    const int wi   = threadIdx.x >> 5;                  // 0..7
    const int m0   = blockIdx.y * 128 + (wi >> 2) * 64; // batch offset
    const int n0   = blockIdx.x * 128 + (wi & 3) * 32;  // hidden offset
    const int lh   = lane & 15;
    const int hi   = lane >> 4;

    const _Float16* abase = z16 + (size_t)(m0 + lh) * DIN + hi * 8;
    const _Float16* bbase = w1h + (size_t)(n0 + lh) * DIN + hi * 8;

    v8f acc[4][2] = {};

#pragma unroll 2
    for (int it = 0; it < DIN / 32; ++it) {
        v16h a[4], b[2];
#pragma unroll
        for (int f = 0; f < 4; ++f) a[f] = load_a_f16(abase + f * (16 * DIN));
#pragma unroll
        for (int g = 0; g < 2; ++g) b[g] = load_b_f16(bbase + g * (16 * DIN));
#pragma unroll
        for (int f = 0; f < 4; ++f)
#pragma unroll
            for (int g = 0; g < 2; ++g)
                acc[f][g] = wmma_f16f32(a[f], b[g], acc[f][g]);
        abase += 32;
        bbase += 32;
    }

#pragma unroll
    for (int f = 0; f < 4; ++f) {
#pragma unroll
        for (int g = 0; g < 2; ++g) {
#pragma unroll
            for (int v = 0; v < 8; ++v) {
                int m = m0 + f * 16 + hi * 8 + v;
                int n = n0 + g * 16 + lh;
                h_out[(size_t)m * HID + n] = acc[f][g][v];
            }
        }
    }
}

// ---------------------------------------------------------------------------
// Kernel B: column mean/var over BATCH (deterministic serial sum per column),
// fold gamma: scale = gamma*rsqrt(var+eps), shift = -mean*scale.
// ---------------------------------------------------------------------------
__global__ void snn_bnstats(const float* __restrict__ h,       // [BATCH, HID]
                            const float* __restrict__ gamma_t, // [HID]
                            float* __restrict__ scale,
                            float* __restrict__ shift)
{
    int col = blockIdx.x * blockDim.x + threadIdx.x;
    float s = 0.0f, s2 = 0.0f;
#pragma unroll 4
    for (int r = 0; r < BATCH; ++r) {
        float v = h[(size_t)r * HID + col];
        s  += v;
        s2 += v * v;
    }
    float mean = s * (1.0f / BATCH);
    float var  = s2 * (1.0f / BATCH) - mean * mean;   // biased, matches torch BN
    float sc   = gamma_t[col] * __frsqrt_rn(var + EPSBN);
    scale[col] = sc;
    shift[col] = -mean * sc;
}

// ---------------------------------------------------------------------------
// Kernel C: BN affine + leaky integrate + threshold/spike/reset (elementwise)
// ---------------------------------------------------------------------------
__global__ void snn_membrane(const float* __restrict__ h,
                             const float* __restrict__ scale,
                             const float* __restrict__ shift,
                             float* __restrict__ mem1,
                             float* __restrict__ spike)
{
    int i = blockIdx.x * blockDim.x + threadIdx.x;   // BATCH*HID
    int col = i & (HID - 1);
    float hn  = h[i] * scale[col] + shift[col];
    float m   = mem1[i] * LEAK + hn;
    float thr = m - 1.0f;           // mem/THRESH - 1, THRESH = 1
    bool  sp  = thr > 0.0f;
    spike[i]  = sp ? 1.0f : 0.0f;
    mem1[i]   = sp ? (m - 1.0f) : m;
}

// ---------------------------------------------------------------------------
// Kernel D: mem2[b,c] += dot(spike[b,:], W2[c,:]); on last step emit out/T.
// ---------------------------------------------------------------------------
__global__ void snn_fc2(const float* __restrict__ spike, // [BATCH, HID]
                        const float* __restrict__ w2,    // [NCLS, HID]
                        float* __restrict__ mem2,        // [BATCH, NCLS]
                        float* __restrict__ out,         // [BATCH, NCLS]
                        int last)
{
    int i = blockIdx.x * blockDim.x + threadIdx.x;
    if (i >= BATCH * NCLS) return;
    int b = i / NCLS, c = i % NCLS;
    const v4f* sr = (const v4f*)(spike + (size_t)b * HID);
    const v4f* wr = (const v4f*)(w2    + (size_t)c * HID);
    float s = 0.0f;
#pragma unroll 4
    for (int k = 0; k < HID / 4; ++k) {
        v4f a = sr[k], w = wr[k];
        s += a.x * w.x + a.y * w.y + a.z * w.z + a.w * w.w;
    }
    float m = mem2[i] + s;
    mem2[i] = m;
    if (last) out[i] = m * (1.0f / TSTEPS);
}

// ---------------------------------------------------------------------------
// Host launcher
// ---------------------------------------------------------------------------
extern "C" void kernel_launch(void* const* d_in, const int* in_sizes, int n_in,
                              void* d_out, int out_size, void* d_ws, size_t ws_size,
                              hipStream_t stream) {
    const float* z     = (const float*)d_in[0];  // [T, B, D]
    const float* W1    = (const float*)d_in[1];  // [H, D]
    const float* gamma = (const float*)d_in[2];  // [T, H]
    const float* W2    = (const float*)d_in[3];  // [C, H]
    float* out = (float*)d_out;                  // [B, C]

    // Workspace carve-up (all offsets 256B aligned); total ~18.3 MB
    char* ws = (char*)d_ws;
    size_t off = 0;
    _Float16* w1h = (_Float16*)(ws + off); off += (size_t)HID * DIN * sizeof(_Float16);   // 8 MB
    _Float16* z16 = (_Float16*)(ws + off); off += (size_t)BATCH * DIN * sizeof(_Float16); // 4 MB
    float* hbuf   = (float*)(ws + off);    off += (size_t)BATCH * HID * sizeof(float);    // 2 MB
    float* spike  = (float*)(ws + off);    off += (size_t)BATCH * HID * sizeof(float);    // 2 MB
    float* mem1   = (float*)(ws + off);    off += (size_t)BATCH * HID * sizeof(float);    // 2 MB
    float* mem2   = (float*)(ws + off);    off += ((size_t)BATCH * NCLS * sizeof(float) + 255) & ~(size_t)255;
    float* scale  = (float*)(ws + off);    off += HID * sizeof(float);
    float* shift  = (float*)(ws + off);    off += HID * sizeof(float);
    (void)ws_size; (void)in_sizes; (void)n_in; (void)out_size;

    // One-time-per-launch prep: W1 -> f16, zero state
    snn_cvt_f32_to_f16<<<(HID * DIN / 4) / 256, 256, 0, stream>>>(W1, w1h);
    snn_zero_state<<<(BATCH * HID + 255) / 256, 256, 0, stream>>>(mem1, mem2);

    dim3 gemmGrid(HID / 128, BATCH / 128);
    for (int t = 0; t < TSTEPS; ++t) {
        const float* z_t = z + (size_t)t * BATCH * DIN;
        snn_cvt_f32_to_f16<<<(BATCH * DIN / 4) / 256, 256, 0, stream>>>(z_t, z16);
        snn_gemm1_wmma<<<gemmGrid, 256, 0, stream>>>(z16, w1h, hbuf);
        snn_bnstats<<<HID / 256, 256, 0, stream>>>(hbuf, gamma + (size_t)t * HID, scale, shift);
        snn_membrane<<<(BATCH * HID) / 256, 256, 0, stream>>>(hbuf, scale, shift, mem1, spike);
        snn_fc2<<<(BATCH * NCLS + 255) / 256, 256, 0, stream>>>(spike, W2, mem2, out,
                                                                t == TSTEPS - 1 ? 1 : 0);
    }
}
